// RelDecoderLayer_52793738002678
// MI455X (gfx1250) — matrix operand order repaired and checked by reading, not compiled, not run on hardware
//
#include <hip/hip_runtime.h>
#include <hip/hip_bf16.h>
#include <math.h>
#include <stdint.h>

// ---------------------------------------------------------------------------
// Types for gfx1250 WMMA (wave32): bf16 16x16x32, f32 accumulate
// ---------------------------------------------------------------------------
typedef __bf16 bf16;
typedef __attribute__((ext_vector_type(16))) bf16  v16bf;
typedef __attribute__((ext_vector_type(8)))  bf16  v8bf;
typedef __attribute__((ext_vector_type(8)))  float v8f;
typedef int v4i __attribute__((vector_size(16)));  // matches builtin pointee

#define WMMA_BF16(a, b, c)                                                   \
  __builtin_amdgcn_wmma_f32_16x16x32_bf16(false, (a), false, (b), (short)0,  \
                                          (c), false, false)

union Frag {
  v16bf v;
  v8bf  h[2];
};

// A fragment (16x32 bf16, M x K): lane l holds row (l&15);
// K slots = {hi*8 .. hi*8+7} U {16+hi*8 .. 16+hi*8+7}, hi = l>>4.
__device__ __forceinline__ v16bf load_a_frag(const bf16* row0, int stride,
                                             int lane) {
  const int m = lane & 15, hi = lane >> 4;
  const bf16* p = row0 + (size_t)m * stride + hi * 8;
  Frag f;
  f.h[0] = *reinterpret_cast<const v8bf*>(p);
  f.h[1] = *reinterpret_cast<const v8bf*>(p + 16);
  return f.v;
}

// B fragment (32x16 bf16, K x N): lane l holds column N=(l&15);
// K slots = {hi*8 .. hi*8+7} U {16+hi*8 .. 16+hi*8+7}, hi = l>>4.
__device__ __forceinline__ v16bf load_b_frag(const bf16* k0, int stride,
                                             int lane) {
  const int n = lane & 15, hi = lane >> 4;
  Frag f;
#pragma unroll
  for (int s = 0; s < 8; ++s) {
    f.h[0][s] = k0[(size_t)(hi * 8 + s) * stride + n];
    f.h[1][s] = k0[(size_t)(16 + hi * 8 + s) * stride + n];
  }
  return f.v;
}

// ---------------------------------------------------------------------------
// CDNA5 async global->LDS copy (16B per lane), ASYNCcnt-tracked.
// Builtin signature (from hipcc diagnostic): param0 = addrspace(1) v4i*,
// then LDS pointer, imm offset, imm cpol.
// ---------------------------------------------------------------------------
__device__ __forceinline__ void async_copy16(const bf16* g, bf16* l) {
#if __has_builtin(__builtin_amdgcn_global_load_async_to_lds_b128)
  __builtin_amdgcn_global_load_async_to_lds_b128(
      (__attribute__((address_space(1))) v4i*)(uintptr_t)g,
      (__attribute__((address_space(3))) v4i*)(uintptr_t)(unsigned)(uintptr_t)l,
      0, 0);
#else
  asm volatile("global_load_async_to_lds_b128 %0, %1, off"
               :
               : "v"((unsigned)(uintptr_t)l),
                 "v"((unsigned long long)(uintptr_t)g)
               : "memory");
#endif
}

__device__ __forceinline__ void wait_async0() {
#if __has_builtin(__builtin_amdgcn_s_wait_asynccnt)
  __builtin_amdgcn_s_wait_asynccnt(0);
#else
  asm volatile("s_wait_asynccnt 0x0" ::: "memory");
#endif
}

// ---------------------------------------------------------------------------
// Conversion / transpose helpers
// ---------------------------------------------------------------------------
__global__ __launch_bounds__(256) void f32_to_bf16_kernel(
    const float* __restrict__ in, bf16* __restrict__ out, int n) {
  int i = blockIdx.x * 256 + threadIdx.x;
  if (i < n) out[i] = (bf16)in[i];
}

// W[N][K] f32 -> Wt[K][N] bf16 (coalesced writes)
__global__ __launch_bounds__(256) void transpose_w_bf16(
    const float* __restrict__ W, bf16* __restrict__ Wt, int N, int K) {
  int i = blockIdx.x * 256 + threadIdx.x;
  if (i >= N * K) return;
  int k = i / N;
  int n = i - k * N;
  Wt[i] = (bf16)W[(size_t)n * K + k];
}

// Kt[bh][d][s] = qkv[(s*2+b)*3072 + 1024 + h*64 + d]   (K transposed per head)
__global__ __launch_bounds__(256) void build_kt_kernel(
    const bf16* __restrict__ qkv, bf16* __restrict__ Kt) {
  int idx = blockIdx.x * 256 + threadIdx.x;  // 32*64*2048 total
  int s  = idx & 2047;
  int d  = (idx >> 11) & 63;
  int bh = idx >> 17;
  int b  = bh >> 4, h = bh & 15;
  Kt[idx] = qkv[(size_t)(s * 2 + b) * 3072 + 1024 + h * 64 + d];
}

// ---------------------------------------------------------------------------
// Tiled WMMA GEMM: out[M][N] = A[M][K](bf16) @ Wt[K][N](bf16)
// 256 threads = 8 waves (4 M-rows x 2 N-cols); wave tile 32x64; block 128x128.
// A/B k-tiles staged in LDS by async global->LDS copies, double-buffered.
// LDS rows padded (A: 40 halfs, B: 136 halfs) to avoid ds_load_b128 bank
// conflicts on the fragment reads.
// OUT_MODE: 0 -> f32 out, 1 -> bf16 out
// ---------------------------------------------------------------------------
#define APAD 40
#define BPAD 136

template <int OUT_MODE>
__global__ __launch_bounds__(256) void gemm_bf16_wmma(
    const bf16* __restrict__ A, const bf16* __restrict__ Wt,
    void* __restrict__ out, int M, int N, int K) {
  (void)M;
  __shared__ __align__(16) bf16 sA[2][128][APAD];
  __shared__ __align__(16) bf16 sB[2][32][BPAD];

  const int tid  = threadIdx.x;
  const int lane = tid & 31;
  const int wave = tid >> 5;
  const int wr = wave >> 1, wc = wave & 1;
  const int bm = blockIdx.y * 128;
  const int bn = blockIdx.x * 128;
  const int m0w = wr * 32;
  const int n0w = wc * 64;

  // Per-thread staging slots: 2 x 16B chunks for A, 2 x 16B chunks for B.
  const int arow = tid >> 2, acol = (tid & 3) * 8;   // rows arow, arow+64
  const int brow = tid >> 4, bcol = (tid & 15) * 8;  // rows brow, brow+16

  auto stage = [&](int buf, int k0) {
    async_copy16(A + (size_t)(bm + arow) * K + k0 + acol,
                 &sA[buf][arow][acol]);
    async_copy16(A + (size_t)(bm + arow + 64) * K + k0 + acol,
                 &sA[buf][arow + 64][acol]);
    async_copy16(Wt + (size_t)(k0 + brow) * N + bn + bcol,
                 &sB[buf][brow][bcol]);
    async_copy16(Wt + (size_t)(k0 + brow + 16) * N + bn + bcol,
                 &sB[buf][brow + 16][bcol]);
  };

  stage(0, 0);
  wait_async0();
  __syncthreads();

  v8f acc[2][4] = {};
  const int nsteps = K >> 5;
  for (int it = 0; it < nsteps; ++it) {
    const int cur = it & 1;
    if (it + 1 < nsteps) stage(cur ^ 1, (it + 1) << 5);

    const v16bf a0 = load_a_frag(&sA[cur][m0w][0], APAD, lane);
    const v16bf a1 = load_a_frag(&sA[cur][m0w + 16][0], APAD, lane);
#pragma unroll
    for (int t = 0; t < 4; ++t) {
      const v16bf b = load_b_frag(&sB[cur][0][n0w + 16 * t], BPAD, lane);
      acc[0][t] = WMMA_BF16(a0, b, acc[0][t]);
      acc[1][t] = WMMA_BF16(a1, b, acc[1][t]);
    }
    wait_async0();     // next tile's async copies (this wave's share) done
    __syncthreads();   // block-wide: tile published, prev buffer free
  }

  const int cn = lane & 15, hi = lane >> 4;
#pragma unroll
  for (int i = 0; i < 2; ++i)
#pragma unroll
    for (int t = 0; t < 4; ++t)
#pragma unroll
      for (int r = 0; r < 8; ++r) {
        const int row = bm + m0w + i * 16 + r + 8 * hi;
        const int col = bn + n0w + 16 * t + cn;
        const float v = acc[i][t][r];
        if (OUT_MODE == 0)
          ((float*)out)[(size_t)row * N + col] = v;
        else
          ((bf16*)out)[(size_t)row * N + col] = (bf16)v;
      }
}

// ---------------------------------------------------------------------------
// Flash attention: one wave handles 16 query rows of one (batch, head).
// Streams keys in blocks of 32 with online softmax; all matmuls via WMMA.
// qkv: bf16 [s][b][3072] (q|k|v each 16 heads x 64)
// Kt : bf16 [b*16+h][64][2048]
// vec: bf16 [s][b][1024] attention output
// ---------------------------------------------------------------------------
__global__ __launch_bounds__(256) void flash_attn_kernel(
    const bf16* __restrict__ qkv, const bf16* __restrict__ Kt,
    bf16* __restrict__ vec) {
  __shared__ float sS[8][16][32];
  __shared__ __align__(16) bf16 sP[8][16][32];

  const int lane = threadIdx.x & 31;
  const int w    = threadIdx.x >> 5;
  const int wid  = blockIdx.x * 8 + w;   // 4096 waves total
  const int iblk = wid & 127;            // 128 query blocks of 16
  const int bh   = wid >> 7;             // 0..31
  const int b    = bh >> 4;
  const int hd   = bh & 15;
  const int i0   = iblk * 16;
  const int mr   = lane & 15;
  const int hi   = lane >> 4;

  // Q fragments for d-chunks [0,32) and [32,64); token row stride = 2*3072
  const bf16* qbase = qkv + (size_t)(i0 * 2 + b) * 3072 + hd * 64;
  const v16bf qa0 = load_a_frag(qbase, 2 * 3072, lane);
  const v16bf qa1 = load_a_frag(qbase + 32, 2 * 3072, lane);
  const bf16* ktbase = Kt + (size_t)bh * 64 * 2048;

  v8f o[4] = {};
  float m_i = -INFINITY, l_i = 0.f;
  const int jmax = i0 + 15;

  for (int j0 = 0; j0 <= jmax; j0 += 32) {
    // ---- scores: two 16x16 tiles (keys j0..j0+31), K-dim = 64 head dims ----
#pragma unroll
    for (int jt = 0; jt < 2; ++jt) {
      v8f s = {};
      const v16bf kb0 =
          load_b_frag(ktbase + j0 + 16 * jt, 2048, lane);
      const v16bf kb1 =
          load_b_frag(ktbase + (size_t)32 * 2048 + j0 + 16 * jt, 2048, lane);
      s = WMMA_BF16(qa0, kb0, s);
      s = WMMA_BF16(qa1, kb1, s);
      const int col = j0 + 16 * jt + mr;  // key index of this lane's column
#pragma unroll
      for (int r = 0; r < 8; ++r) {
        const int row = i0 + r + 8 * hi;  // query index of this element
        sS[w][r + 8 * hi][16 * jt + mr] =
            (col <= row) ? s[r] * 0.125f : -INFINITY;
      }
    }
    __builtin_amdgcn_wave_barrier();

    // ---- online softmax: lane handles row mr, key half hi ----
    float vals[16];
    float mx = -INFINITY;
#pragma unroll
    for (int c = 0; c < 16; ++c) {
      vals[c] = sS[w][mr][hi * 16 + c];
      mx = fmaxf(mx, vals[c]);
    }
    mx = fmaxf(mx, __shfl_xor(mx, 16, 32));
    const float m_new = fmaxf(m_i, mx);
    const float alpha = __expf(m_i - m_new);
    float lsum = 0.f;
#pragma unroll
    for (int c = 0; c < 16; ++c) {
      const float p = __expf(vals[c] - m_new);
      lsum += p;
      sP[w][mr][hi * 16 + c] = (bf16)p;
    }
    lsum += __shfl_xor(lsum, 16, 32);
    l_i = l_i * alpha + lsum;
    m_i = m_new;

    // rescale accumulators: element row = r + 8*hi, stats live in lane (row)
    float af[8];
#pragma unroll
    for (int r = 0; r < 8; ++r) af[r] = __shfl(alpha, r + 8 * hi, 32);
#pragma unroll
    for (int t = 0; t < 4; ++t)
#pragma unroll
      for (int r = 0; r < 8; ++r) o[t][r] *= af[r];

    __builtin_amdgcn_wave_barrier();

    // ---- P @ V : A = P[16x32] from LDS, B = V[32x16] natural layout ----
    Frag pf;
    pf.h[0] = *reinterpret_cast<const v8bf*>(&sP[w][mr][hi * 8]);
    pf.h[1] = *reinterpret_cast<const v8bf*>(&sP[w][mr][16 + hi * 8]);
    const bf16* vbase =
        qkv + (size_t)(j0 * 2 + b) * 3072 + 2048 + hd * 64;
#pragma unroll
    for (int t = 0; t < 4; ++t) {
      const v16bf vb = load_b_frag(vbase + 16 * t, 2 * 3072, lane);
      o[t] = WMMA_BF16(pf.v, vb, o[t]);
    }
    __builtin_amdgcn_wave_barrier();  // sS/sP reused next iteration
  }

  const float linv = 1.f / l_i;
  float lf[8];
#pragma unroll
  for (int r = 0; r < 8; ++r) lf[r] = __shfl(linv, r + 8 * hi, 32);
#pragma unroll
  for (int t = 0; t < 4; ++t)
#pragma unroll
    for (int r = 0; r < 8; ++r) {
      const int srow = i0 + r + 8 * hi;
      vec[(size_t)(srow * 2 + b) * 1024 + hd * 64 + 16 * t + mr] =
          (bf16)(o[t][r] * lf[r]);
    }
}

// ---------------------------------------------------------------------------
// Fused residual add + LayerNorm over 1024 features; one block per token.
// Writes f32 result; optionally a bf16 copy (GEMM A operand for the FFN).
// ---------------------------------------------------------------------------
__global__ __launch_bounds__(256) void add_layernorm_kernel(
    const float* __restrict__ X, const float* __restrict__ Y,
    const float* __restrict__ G, const float* __restrict__ Bt,
    float* __restrict__ Of, bf16* __restrict__ Ob) {
  __shared__ float red1[8], red2[8];
  const int t = threadIdx.x, lane = t & 31, w = t >> 5;
  const size_t base = (size_t)blockIdx.x * 1024;

  const float4 xv = *reinterpret_cast<const float4*>(X + base + t * 4);
  const float4 yv = *reinterpret_cast<const float4*>(Y + base + t * 4);
  const float v0 = xv.x + yv.x, v1 = xv.y + yv.y;
  const float v2 = xv.z + yv.z, v3 = xv.w + yv.w;

  float s = v0 + v1 + v2 + v3;
#pragma unroll
  for (int off = 16; off; off >>= 1) s += __shfl_xor(s, off, 32);
  if (lane == 0) red1[w] = s;
  __syncthreads();
  float tot = 0.f;
#pragma unroll
  for (int i = 0; i < 8; ++i) tot += red1[i];
  const float mean = tot * (1.f / 1024.f);

  const float d0 = v0 - mean, d1 = v1 - mean, d2 = v2 - mean, d3 = v3 - mean;
  float q = d0 * d0 + d1 * d1 + d2 * d2 + d3 * d3;
#pragma unroll
  for (int off = 16; off; off >>= 1) q += __shfl_xor(q, off, 32);
  if (lane == 0) red2[w] = q;
  __syncthreads();
  float qt = 0.f;
#pragma unroll
  for (int i = 0; i < 8; ++i) qt += red2[i];
  const float rstd = rsqrtf(qt * (1.f / 1024.f) + 1e-5f);

  const float4 gv = *reinterpret_cast<const float4*>(G + t * 4);
  const float4 bv = *reinterpret_cast<const float4*>(Bt + t * 4);
  const float o0 = d0 * rstd * gv.x + bv.x;
  const float o1 = d1 * rstd * gv.y + bv.y;
  const float o2 = d2 * rstd * gv.z + bv.z;
  const float o3 = d3 * rstd * gv.w + bv.w;
  *reinterpret_cast<float4*>(Of + base + t * 4) = make_float4(o0, o1, o2, o3);
  if (Ob) {
    Ob[base + t * 4 + 0] = (bf16)o0;
    Ob[base + t * 4 + 1] = (bf16)o1;
    Ob[base + t * 4 + 2] = (bf16)o2;
    Ob[base + t * 4 + 3] = (bf16)o3;
  }
}

// ---------------------------------------------------------------------------
// Launcher. Workspace layout (bytes), total 104 MiB with region reuse:
//   [0      ) qkv_wt  6 MiB | [6 MiB ) o_wt 2 MiB | [8 MiB ) w1t 8 MiB
//   [16 MiB ) w2t 8 MiB     | [24 MiB) h16 8 MiB  | [32 MiB) qkv 24 MiB
//   [56 MiB ) Kt 8 MiB      | [64 MiB) vec 8 MiB  | [72 MiB) attn_out 16 MiB
//   [88 MiB ) x 16 MiB      | l1 reuses qkv+Kt, x16 reuses vec, l2 reuses attn
// ---------------------------------------------------------------------------
extern "C" void kernel_launch(void* const* d_in, const int* in_sizes, int n_in,
                              void* d_out, int out_size, void* d_ws,
                              size_t ws_size, hipStream_t stream) {
  (void)in_sizes; (void)n_in; (void)out_size; (void)ws_size;
  const float* h     = (const float*)d_in[0];
  // d_in[1] = pos_emb (unused by reference), d_in[2] = attn_mask (causal)
  const float* qkv_w = (const float*)d_in[3];
  const float* o_w   = (const float*)d_in[4];
  const float* ln1_g = (const float*)d_in[5];
  const float* ln1_b = (const float*)d_in[6];
  const float* w1    = (const float*)d_in[7];
  const float* w2    = (const float*)d_in[8];
  const float* ln2_g = (const float*)d_in[9];
  const float* ln2_b = (const float*)d_in[10];

  char* ws = (char*)d_ws;
  bf16*  qkv_wt   = (bf16*)(ws + 0);
  bf16*  o_wt     = (bf16*)(ws + 6291456);
  bf16*  w1t      = (bf16*)(ws + 8388608);
  bf16*  w2t      = (bf16*)(ws + 16777216);
  bf16*  h16      = (bf16*)(ws + 25165824);
  bf16*  qkvb     = (bf16*)(ws + 33554432);
  bf16*  Kt       = (bf16*)(ws + 58720256);
  bf16*  vec      = (bf16*)(ws + 67108864);
  float* attn_out = (float*)(ws + 75497472);
  float* xbuf     = (float*)(ws + 92274688);
  bf16*  l1       = (bf16*)(ws + 33554432);   // reuses qkv+Kt (dead)
  bf16*  x16      = (bf16*)(ws + 67108864);   // reuses vec (dead)
  float* l2       = (float*)(ws + 75497472);  // reuses attn_out (dead)

  // 1) bf16 weight transposes + activation convert
  transpose_w_bf16<<<12288, 256, 0, stream>>>(qkv_w, qkv_wt, 3072, 1024);
  transpose_w_bf16<<<4096,  256, 0, stream>>>(o_w,   o_wt,   1024, 1024);
  transpose_w_bf16<<<16384, 256, 0, stream>>>(w1,    w1t,    4096, 1024);
  transpose_w_bf16<<<16384, 256, 0, stream>>>(w2,    w2t,    1024, 4096);
  f32_to_bf16_kernel<<<16384, 256, 0, stream>>>(h, h16, 4194304);

  // 2) QKV projection
  gemm_bf16_wmma<1><<<dim3(24, 32), 256, 0, stream>>>(
      h16, qkv_wt, (void*)qkvb, 4096, 3072, 1024);

  // 3) K transpose per head, then flash attention
  build_kt_kernel<<<16384, 256, 0, stream>>>(qkvb, Kt);
  flash_attn_kernel<<<512, 256, 0, stream>>>(qkvb, Kt, vec);

  // 4) output projection + residual LN1 (writes fp32 x and bf16 x16)
  gemm_bf16_wmma<0><<<dim3(8, 32), 256, 0, stream>>>(
      vec, o_wt, (void*)attn_out, 4096, 1024, 1024);
  add_layernorm_kernel<<<4096, 256, 0, stream>>>(h, attn_out, ln1_g, ln1_b,
                                                 xbuf, x16);

  // 5) FFN
  gemm_bf16_wmma<1><<<dim3(32, 32), 256, 0, stream>>>(
      x16, w1t, (void*)l1, 4096, 4096, 1024);
  gemm_bf16_wmma<0><<<dim3(8, 32), 256, 0, stream>>>(
      l1, w2t, (void*)l2, 4096, 1024, 4096);

  // 6) residual + LN2 -> final fp32 output
  add_layernorm_kernel<<<4096, 256, 0, stream>>>(xbuf, l2, ln2_g, ln2_b,
                                                 (float*)d_out, nullptr);
}
